// Correlation_74277164417490
// MI455X (gfx1250) — matrix operand order repaired and checked by reading, not compiled, not run on hardware
//
#include <hip/hip_runtime.h>

typedef __attribute__((ext_vector_type(16))) __bf16 v16bf;
typedef __attribute__((ext_vector_type(8)))  float  v8f;
typedef __attribute__((address_space(3))) unsigned short lds_us;

#define C_DIM 1024
#define S_DIM 1024
#define N_BATCH 32
#define BTILES 32           // C_DIM / 32  (32x32 output blocks)
#define TRI_TILES 528       // BTILES*(BTILES+1)/2
#define WAVES_PER_BLOCK 4
#define BUF_BYTES 2048      // one 32x32 bf16 tile

__device__ __forceinline__ unsigned short f2bf(float f) {
    // round-to-nearest-even float -> bf16
    unsigned int u = __float_as_uint(f);
    unsigned int r = (u + 0x7FFFu + ((u >> 16) & 1u)) >> 16;
    return (unsigned short)r;
}

// gfx1250 async DMA: global -> LDS, 16B per lane, tracked by ASYNCcnt.
// (inline asm: builtin arity differs across toolchains)
__device__ __forceinline__ void async_load_b128(unsigned lds_off,
                                                const unsigned short* gptr) {
    asm volatile("global_load_async_to_lds_b128 %0, %1, off"
                 :: "v"(lds_off),
                    "v"((unsigned long long)(size_t)gptr)
                 : "memory");
}
__device__ __forceinline__ void wait_async0() {
    asm volatile("s_wait_asynccnt 0x0" ::: "memory");
}
__device__ __forceinline__ unsigned lds_offset(const unsigned short* p) {
    return (unsigned)(size_t)(lds_us*)p;   // addrspace(3) ptr == byte offset
}

// ---------------------------------------------------------------------------
// Kernel 1: per (n,c) row over S: mean-center, unit-norm, emit bf16
// ---------------------------------------------------------------------------
__global__ void normalize_rows(const float* __restrict__ x,
                               unsigned short* __restrict__ xn) {
    __shared__ float red[256];
    const int row = blockIdx.x;                 // n*C + c
    const int tid = threadIdx.x;                // 256 threads, 4 floats each
    const float4* xr = (const float4*)(x + (size_t)row * S_DIM);
    float4 v = xr[tid];

    float s = v.x + v.y + v.z + v.w;
    red[tid] = s; __syncthreads();
    for (int off = 128; off > 0; off >>= 1) {
        if (tid < off) red[tid] += red[tid + off];
        __syncthreads();
    }
    const float mean = red[0] * (1.0f / (float)S_DIM);
    __syncthreads();

    float4 c;
    c.x = v.x - mean; c.y = v.y - mean; c.z = v.z - mean; c.w = v.w - mean;
    float sq = c.x * c.x + c.y * c.y + c.z * c.z + c.w * c.w;
    red[tid] = sq; __syncthreads();
    for (int off = 128; off > 0; off >>= 1) {
        if (tid < off) red[tid] += red[tid + off];
        __syncthreads();
    }
    const float inv = 1.0f / (sqrtf(red[0]) + 1e-8f);

    ushort4 o;
    o.x = f2bf(c.x * inv); o.y = f2bf(c.y * inv);
    o.z = f2bf(c.z * inv); o.w = f2bf(c.w * inv);
    *(ushort4*)(xn + (size_t)row * S_DIM + (size_t)tid * 4) = o;
}

// ---------------------------------------------------------------------------
// Kernel 2: one wave per 32x32 tile of corr = Xn * Xn^T (upper triangle only).
// Staging is async DMA (global_load_async_to_lds_b128) into double-buffered
// wave-private LDS; next chunk's DMA overlaps this chunk's 4 WMMAs and one
// s_wait_asynccnt 0 per K-step closes the pipeline. No block barriers.
// ---------------------------------------------------------------------------
__global__ void corr_tiles(const unsigned short* __restrict__ xn,
                           float* __restrict__ partial) {
    __shared__ unsigned short ldsA[WAVES_PER_BLOCK][2][32 * 32];  // 4 KB/wave
    __shared__ unsigned short ldsB[WAVES_PER_BLOCK][2][32 * 32];

    const int wave = threadIdx.x >> 5;
    const int lane = threadIdx.x & 31;
    const int n    = blockIdx.y;
    const int t    = blockIdx.x * WAVES_PER_BLOCK + wave;   // triangular tile id

    // decode t -> (i, j), j >= i
    int i = 0, rem = t, rowlen = BTILES;
    while (rem >= rowlen) { rem -= rowlen; ++i; --rowlen; }
    const int j = i + rem;

    const unsigned short* base = xn + ((size_t)n << 20);          // n*C*S
    const unsigned short* gA = base + (size_t)(i * 32) * S_DIM;   // 32 rows, tile i
    const unsigned short* gB = base + (size_t)(j * 32) * S_DIM;   // 32 rows, tile j

    const int half = lane >> 4;
    const int mn   = lane & 15;

    // staging: 32 rows x 32 bf16 = 2 KB = 128 x 16B chunks; lane does 4 chunks
    const unsigned short* gAq[4];
    const unsigned short* gBq[4];
    unsigned offA[4], offB[4];          // LDS byte offsets (buffer 0)
    #pragma unroll
    for (int q = 0; q < 4; ++q) {
        const int ch  = lane + 32 * q;
        const int row = ch >> 2;
        const int off = (ch & 3) * 8;
        gAq[q] = gA + (size_t)row * S_DIM + off;
        gBq[q] = gB + (size_t)row * S_DIM + off;
        offA[q] = lds_offset(&ldsA[wave][0][row * 32 + off]);
        offB[q] = lds_offset(&ldsB[wave][0][row * 32 + off]);
    }

    // prologue: DMA chunk 0 into buffer 0
    #pragma unroll
    for (int q = 0; q < 4; ++q) {
        async_load_b128(offA[q], gAq[q]);
        async_load_b128(offB[q], gBq[q]);
    }
    wait_async0();

    v8f acc00 = {}, acc01 = {}, acc10 = {}, acc11 = {};

    for (int ks = 0; ks < S_DIM; ks += 32) {
        const int pb = (ks >> 5) & 1;            // current LDS buffer
        if (ks + 32 < S_DIM) {                   // DMA next chunk into other buffer
            const unsigned nb = (unsigned)(pb ^ 1) * BUF_BYTES;
            #pragma unroll
            for (int q = 0; q < 4; ++q) {
                async_load_b128(offA[q] + nb, gAq[q] + ks + 32);
                async_load_b128(offB[q] + nb, gBq[q] + ks + 32);
            }
        }

        // A fragment (16x32 MxK): lane row M=mn(+16); VGPR v -> K pair at
        //   ka = (v/4)*16 + half*8 + (v%4)*2
        // B fragment (32x16 KxN): lane col N=mn(+16); VGPR v -> K pair at
        //   kb = (v/4)*16 + half*8 + (v%4)*2   (B[k][n] = row n of tile j at col k)
        union { unsigned int u[8]; v16bf v; } a0, a1, b0, b1;
        #pragma unroll
        for (int vv = 0; vv < 8; ++vv) {
            const int ka = ((vv >> 2) * 16) + half * 8 + (vv & 3) * 2;
            const int kb = ((vv >> 2) * 16) + half * 8 + (vv & 3) * 2;
            a0.u[vv] = *(const unsigned int*)&ldsA[wave][pb][mn * 32 + ka];
            a1.u[vv] = *(const unsigned int*)&ldsA[wave][pb][(mn + 16) * 32 + ka];
            b0.u[vv] = *(const unsigned int*)&ldsB[wave][pb][mn * 32 + kb];
            b1.u[vv] = *(const unsigned int*)&ldsB[wave][pb][(mn + 16) * 32 + kb];
        }
        acc00 = __builtin_amdgcn_wmma_f32_16x16x32_bf16(false, a0.v, false, b0.v,
                                                        (short)0, acc00, false, false);
        acc01 = __builtin_amdgcn_wmma_f32_16x16x32_bf16(false, a0.v, false, b1.v,
                                                        (short)0, acc01, false, false);
        acc10 = __builtin_amdgcn_wmma_f32_16x16x32_bf16(false, a1.v, false, b0.v,
                                                        (short)0, acc10, false, false);
        acc11 = __builtin_amdgcn_wmma_f32_16x16x32_bf16(false, a1.v, false, b1.v,
                                                        (short)0, acc11, false, false);

        if (ks + 32 < S_DIM)
            wait_async0();                       // next buffer ready before swap
    }

    // C/D layout: element (M = r + 8*half, N = mn) in acc[r]
    // Global coords: acc00:(i*32+M,     j*32+N)      acc01:(i*32+M,     j*32+16+N)
    //                acc10:(i*32+16+M,  j*32+N)      acc11:(i*32+16+M,  j*32+16+N)
    // For i==j only acc00/acc11 can touch the true diagonal (M==N).
    float s = 0.0f;
    if (i == j) {
        #pragma unroll
        for (int r = 0; r < 8; ++r) {
            const int M = r + half * 8;
            const float wd = (M == mn) ? 0.0f : 1.0f;
            s += wd * fabsf(acc00[r]) + fabsf(acc01[r]) +
                 fabsf(acc10[r]) + wd * fabsf(acc11[r]);
        }
    } else {
        #pragma unroll
        for (int r = 0; r < 8; ++r)
            s += fabsf(acc00[r]) + fabsf(acc01[r]) +
                 fabsf(acc10[r]) + fabsf(acc11[r]);
        s *= 2.0f;   // symmetric partner block
    }
    for (int off = 16; off > 0; off >>= 1)
        s += __shfl_down(s, off, 32);
    if (lane == 0)
        partial[(size_t)n * TRI_TILES + t] = s;
}

// ---------------------------------------------------------------------------
// Kernel 3: deterministic single-block reduction + final scaling
// ---------------------------------------------------------------------------
__global__ void reduce_scale(const float* __restrict__ partial,
                             float* __restrict__ out) {
    __shared__ float red[256];
    const int tid = threadIdx.x;
    float s = 0.0f;
    const int n = N_BATCH * TRI_TILES;
    for (int idx = tid; idx < n; idx += 256) s += partial[idx];
    red[tid] = s; __syncthreads();
    for (int off = 128; off > 0; off >>= 1) {
        if (tid < off) red[tid] += red[tid + off];
        __syncthreads();
    }
    if (tid == 0) {
        const float num_comb = (float)((C_DIM * (C_DIM - 1)) / 2);
        out[0] = red[0] / num_comb / 2.0f / (float)N_BATCH;
    }
}

extern "C" void kernel_launch(void* const* d_in, const int* in_sizes, int n_in,
                              void* d_out, int out_size, void* d_ws, size_t ws_size,
                              hipStream_t stream) {
    (void)in_sizes; (void)n_in; (void)out_size; (void)ws_size;
    const float* x = (const float*)d_in[0];
    unsigned short* xn = (unsigned short*)d_ws;                       // 64 MB bf16
    float* partial = (float*)((char*)d_ws +
                              (size_t)N_BATCH * C_DIM * S_DIM * sizeof(unsigned short));
    float* out = (float*)d_out;

    normalize_rows<<<dim3(N_BATCH * C_DIM), dim3(256), 0, stream>>>(x, xn);
    corr_tiles<<<dim3(TRI_TILES / WAVES_PER_BLOCK, N_BATCH),
                 dim3(WAVES_PER_BLOCK * 32), 0, stream>>>(xn, partial);
    reduce_scale<<<dim3(1), dim3(256), 0, stream>>>(partial, out);
}
